// Model_68186900792086
// MI455X (gfx1250) — matrix operand order repaired and checked by reading, not compiled, not run on hardware
//
#include <hip/hip_runtime.h>
#include <hip/hip_bf16.h>
#include <math.h>

#define Bsz   8
#define Tsz   4096
#define Hsz   4
#define HVsz  8
#define Ksz   128
#define Vsz   128
#define CHUNK 16
#define VT    16   // V-columns per wave

typedef __attribute__((ext_vector_type(16))) __bf16 v16bf;
typedef __attribute__((ext_vector_type(8)))  float  v8f;

__device__ __forceinline__ float softplus_f(float x) {
    return (x <= 20.f) ? log1pf(expf(x)) : x;
}
__device__ __forceinline__ float sigmoid_f(float x) { return 1.f / (1.f + expf(-x)); }

// ---- CDNA5 WMMA operand loaders (wave32 layouts per ISA 7.12.2) ----
// A matrix 16x32 bf16, row-major source [m][k] with leading dim ld.
__device__ __forceinline__ v16bf ld_a(const __bf16* base, int ld, int kc, int lane) {
    int m = lane & 15, g = (lane >> 4) & 1;
    v16bf r;
#pragma unroll
    for (int e = 0; e < 16; ++e) {
        int kk = (e < 8) ? (8 * g + e) : (8 * g + e + 8);
        r[e] = base[m * ld + kc + kk];
    }
    return r;
}
// A with per-row f32 scale (decay D_t applied at load).
__device__ __forceinline__ v16bf ld_a_scaled(const __bf16* base, int ld, int kc, int lane,
                                             const float* rowScale) {
    int m = lane & 15, g = (lane >> 4) & 1;
    float s = rowScale[m];
    v16bf r;
#pragma unroll
    for (int e = 0; e < 16; ++e) {
        int kk = (e < 8) ? (8 * g + e) : (8 * g + e + 8);
        r[e] = (__bf16)(s * (float)base[m * ld + kc + kk]);
    }
    return r;
}
// A = lower-triangular 16x16 f32 matrix, zero-padded to K=32.
__device__ __forceinline__ v16bf ld_a_tril(const float* S, int lane) {
    int m = lane & 15, g = (lane >> 4) & 1;
    v16bf r;
#pragma unroll
    for (int e = 0; e < 16; ++e) {
        int kk = (e < 8) ? (8 * g + e) : (8 * g + e + 8);
        float v = (kk < 16 && kk <= m) ? S[m * 16 + kk] : 0.f;
        r[e] = (__bf16)v;
    }
    return r;
}
// B matrix 32x16 bf16; source stored "column-major" [n][k] (contraction contiguous).
__device__ __forceinline__ v16bf ld_b(const __bf16* base, int ld, int kc, int lane) {
    int n = lane & 15, g = (lane >> 4) & 1;
    v16bf r;
#pragma unroll
    for (int e = 0; e < 16; ++e) r[e] = base[n * ld + kc + ((e < 8) ? (8 * g + e) : (8 * g + e + 8))];
    return r;
}
__device__ __forceinline__ v8f wmma_bf16(v16bf a, v16bf b, v8f c) {
    return __builtin_amdgcn_wmma_f32_16x16x32_bf16(false, a, false, b, (short)0, c, false, false);
}

__global__ __launch_bounds__(256) void gdn_chunked_kernel(
    const float* __restrict__ A_log, const float* __restrict__ a_in,
    const float* __restrict__ dt_bias, const float* __restrict__ q_in,
    const float* __restrict__ k_in, const float* __restrict__ v_in,
    const float* __restrict__ b_in, const float* __restrict__ h0_src,
    const int* __restrict__ h0_idx, float* __restrict__ out) {
    __shared__ __bf16 hbf[Vsz * Ksz];    // state bf16 copy, [v][k]
    __shared__ __bf16 Qn[CHUNK * Ksz];   // l2norm(q)*scale, [t][k]
    __shared__ __bf16 Kn[CHUNK * Ksz];   // l2norm(k),       [t][k]
    __shared__ __bf16 KcT[Ksz * 32];     // exp(Gtot-G_s)*k_s, [k][s], s padded to 32
    __shared__ __bf16 Ubf[Vsz * 32];     // RHS then U, [v][s], s padded to 32
    __shared__ float  Ssc[16 * 16];      // tril_incl(Q K^T * exp(G_t-G_s))
    __shared__ float  Lsc[16 * 16];      // strict-lower beta_t*(K K^T)*exp(G_t-G_s)
    __shared__ float  sg[16], sG[16], sD[16], sKrel[16], sbeta[16];
    __shared__ float  sDtot;

    const int tid  = threadIdx.x;
    const int lane = tid & 31;
    const int wave = tid >> 5;
    const int bb   = blockIdx.x / HVsz;
    const int hv   = blockIdx.x % HVsz;
    const int hq   = hv / (HVsz / Hsz);   // GQA head
    const int vbase = wave * VT;
    const int g16  = (lane >> 4) & 1;
    const int nl   = lane & 15;
    const float qscale = 0.08838834764831845f;  // K^-0.5
    const float negexpA = -expf(A_log[hv]);
    const float dtb = dt_bias[hv];

    // zero the K-padding columns (never touched again)
    for (int i = tid; i < Ksz * 16; i += 256) KcT[(i >> 4) * 32 + 16 + (i & 15)] = (__bf16)0.f;
    for (int i = tid; i < Vsz * 16; i += 256) Ubf[(i >> 4) * 32 + 16 + (i & 15)] = (__bf16)0.f;

    // initial state: registers (f32, WMMA C/D layout) + bf16 LDS copy
    const int idx = h0_idx[bb];
    const float* hsrc = (idx >= 0) ? (h0_src + ((size_t)idx * HVsz + hv) * Ksz * Vsz) : nullptr;
    v8f hst[8];
#pragma unroll
    for (int m = 0; m < 8; ++m) {
#pragma unroll
        for (int j = 0; j < 8; ++j) {
            int kk = 16 * m + 8 * g16 + j;
            hst[m][j] = hsrc ? hsrc[kk * Vsz + vbase + nl] : 0.f;
        }
    }
    for (int e = lane; e < VT * Ksz; e += 32) {
        int vv = vbase + (e >> 7), kk = e & 127;
        hbf[vv * Ksz + kk] = (__bf16)(hsrc ? hsrc[kk * Vsz + vv] : 0.f);
    }
    __syncthreads();

    for (int c = 0; c < Tsz / CHUNK; ++c) {
        const int t0 = c * CHUNK;
        // --- phase 1a: per-step gates ---
        if (tid < 16) {
            size_t off = ((size_t)bb * Tsz + t0 + tid) * HVsz + hv;
            sg[tid]    = negexpA * softplus_f(a_in[off] + dtb);
            sbeta[tid] = sigmoid_f(b_in[off]);
        }
        __syncthreads();
        if (tid == 0) {
            float acc = 0.f;
            for (int t = 0; t < 16; ++t) { acc += sg[t]; sG[t] = acc; sD[t] = expf(acc); }
            for (int t = 0; t < 16; ++t) sKrel[t] = expf(sG[15] - sG[t]);
            sDtot = sD[15];
        }
        __syncthreads();
        // --- phase 1c: l2norm q/k rows, stage bf16 operands (2 rows per wave) ---
#pragma unroll
        for (int rr = 0; rr < 2; ++rr) {
            int t = wave * 2 + rr;
            size_t qoff = (((size_t)bb * Tsz + t0 + t) * Hsz + hq) * (size_t)Ksz;
            float qv[4], kv[4], sq = 0.f, sk = 0.f;
#pragma unroll
            for (int i = 0; i < 4; ++i) {
                int kk = i * 32 + lane;
                qv[i] = q_in[qoff + kk]; kv[i] = k_in[qoff + kk];
                sq += qv[i] * qv[i]; sk += kv[i] * kv[i];
            }
#pragma unroll
            for (int m = 16; m >= 1; m >>= 1) { sq += __shfl_xor(sq, m, 32); sk += __shfl_xor(sk, m, 32); }
            float rq = qscale * rsqrtf(sq + 1e-6f);
            float rk = rsqrtf(sk + 1e-6f);
            float krel = sKrel[t];
#pragma unroll
            for (int i = 0; i < 4; ++i) {
                int kk = i * 32 + lane;
                float knv = kv[i] * rk;
                Qn[t * Ksz + kk]  = (__bf16)(qv[i] * rq);
                Kn[t * Ksz + kk]  = (__bf16)knv;
                KcT[kk * 32 + t]  = (__bf16)(krel * knv);
            }
            if (c + 1 < Tsz / CHUNK) {  // prefetch next chunk (global_prefetch_b8)
                __builtin_prefetch(&q_in[qoff + (size_t)CHUNK * Hsz * Ksz + lane * 4], 0, 1);
                __builtin_prefetch(&k_in[qoff + (size_t)CHUNK * Hsz * Ksz + lane * 4], 0, 1);
            }
        }
        __syncthreads();
        // --- phase 2: chunk GEMMs against state ---
        v8f sqk = {0,0,0,0,0,0,0,0}, skk = {0,0,0,0,0,0,0,0};
        v8f wac = {0,0,0,0,0,0,0,0}, oac = {0,0,0,0,0,0,0,0};
#pragma unroll
        for (int kc2 = 0; kc2 < Ksz; kc2 += 32) {
            v16bf bk  = ld_b(Kn, Ksz, kc2, lane);
            sqk = wmma_bf16(ld_a(Qn, Ksz, kc2, lane), bk, sqk);          // Q K^T
            skk = wmma_bf16(ld_a(Kn, Ksz, kc2, lane), bk, skk);          // K K^T
            v16bf bh0 = ld_b(&hbf[vbase * Ksz], Ksz, kc2, lane);
            wac = wmma_bf16(ld_a_scaled(Kn, Ksz, kc2, lane, sD), bh0, wac);  // (D.K) h0
            oac = wmma_bf16(ld_a_scaled(Qn, Ksz, kc2, lane, sD), bh0, oac);  // (D.Q) h0
        }
        // RHS_t[v] = beta_t * (v_t[v] - W[t][v])  -> Ubf (bf16)
#pragma unroll
        for (int j = 0; j < 8; ++j) {
            int t = 8 * g16 + j, vv = vbase + nl;
            float vval = v_in[(((size_t)bb * Tsz + t0 + t) * HVsz + hv) * Vsz + vv];
            Ubf[vv * 32 + t] = (__bf16)(sbeta[t] * (vval - wac[j]));
        }
        if (wave == 0) {  // decay-masked score matrices
#pragma unroll
            for (int j = 0; j < 8; ++j) {
                int t = 8 * g16 + j, s = nl;
                float p = expf(sG[t] - sG[s]);
                Ssc[t * 16 + s] = (s <= t) ? sqk[j] * p : 0.f;
                Lsc[t * 16 + s] = (s <  t) ? sbeta[t] * skk[j] * p : 0.f;
            }
        }
        __syncthreads();
        // --- phase 3: forward substitution (I+L) U = RHS; one thread per V column ---
        if (tid < Vsz) {
            int vv = tid;
            float u[16];
#pragma unroll
            for (int t = 0; t < 16; ++t) {
                float acc = (float)Ubf[vv * 32 + t];
#pragma unroll
                for (int s = 0; s < t; ++s) acc -= Lsc[t * 16 + s] * u[s];
                u[t] = acc;
                Ubf[vv * 32 + t] = (__bf16)acc;
            }
        }
        __syncthreads();
        // --- phase 4: intra-chunk output + state update ---
        v16bf bu = ld_b(&Ubf[vbase * 32], 32, 0, lane);
        oac = wmma_bf16(ld_a_tril(Ssc, lane), bu, oac);   // O += tril(S) U
#pragma unroll
        for (int j = 0; j < 8; ++j) {
            int t = 8 * g16 + j, vv = vbase + nl;
            out[(((size_t)bb * Tsz + t0 + t) * HVsz + hv) * Vsz + vv] = oac[j];
        }
        float dtot = sDtot;
#pragma unroll
        for (int m = 0; m < 8; ++m) {  // h = Dtot*h + Kc^T U  (8 stacked 16x16 tiles)
#pragma unroll
            for (int j = 0; j < 8; ++j) hst[m][j] *= dtot;
            hst[m] = wmma_bf16(ld_a(&KcT[(16 * m) * 32], 32, 0, lane), bu, hst[m]);
        }
#pragma unroll
        for (int m = 0; m < 8; ++m) {  // refresh bf16 state copy for next chunk
#pragma unroll
            for (int j = 0; j < 8; ++j) {
                int kk = 16 * m + 8 * g16 + j;
                hbf[(vbase + nl) * Ksz + kk] = (__bf16)hst[m][j];
            }
        }
        __syncthreads();
    }
}

extern "C" void kernel_launch(void* const* d_in, const int* in_sizes, int n_in,
                              void* d_out, int out_size, void* d_ws, size_t ws_size,
                              hipStream_t stream) {
    (void)in_sizes; (void)n_in; (void)out_size; (void)d_ws; (void)ws_size;
    const float* A_log = (const float*)d_in[0];
    const float* a     = (const float*)d_in[1];
    const float* dtb   = (const float*)d_in[2];
    const float* q     = (const float*)d_in[3];
    const float* k     = (const float*)d_in[4];
    const float* v     = (const float*)d_in[5];
    const float* b     = (const float*)d_in[6];
    const float* hsrc  = (const float*)d_in[7];
    const int*   hidx  = (const int*)d_in[8];
    float* out = (float*)d_out;
    hipLaunchKernelGGL(gdn_chunked_kernel, dim3(Bsz * HVsz), dim3(256), 0, stream,
                       A_log, a, dtb, q, k, v, b, hsrc, hidx, out);
}
